// RotaryMultiheadAttention_56959856280328
// MI455X (gfx1250) — matrix operand order repaired and checked by reading, not compiled, not run on hardware
//
#include <hip/hip_runtime.h>
#include <hip/hip_bf16.h>

typedef __attribute__((ext_vector_type(16))) __bf16          bf16x16;
typedef __attribute__((ext_vector_type(8)))  float           floatx8;
typedef __attribute__((ext_vector_type(16))) unsigned short  ushort16;
typedef __attribute__((ext_vector_type(8)))  unsigned short  ushort8;

#define SEQ    2048
#define BATCH  2
#define DMODEL 1024
#define NHEAD  16
#define HD     64
#define N3     3072
#define SB     (SEQ * BATCH)   // 4096 rows

__device__ __forceinline__ unsigned short f2bf(float f) {
    unsigned int u = __float_as_uint(f);
    u += 0x7FFFu + ((u >> 16) & 1u);   // round-to-nearest-even
    return (unsigned short)(u >> 16);
}
__device__ __forceinline__ float bf2f(unsigned short h) {
    return __uint_as_float(((unsigned int)h) << 16);
}
__device__ __forceinline__ ushort8 ld8(const unsigned short* p) {
    return *(const ushort8*)p;           // global_load_b128
}
__device__ __forceinline__ bf16x16 frag(ushort8 lo, ushort8 hi) {
    ushort16 u = __builtin_shufflevector(lo, hi, 0, 1, 2, 3, 4, 5, 6, 7,
                                                 8, 9, 10, 11, 12, 13, 14, 15);
    return __builtin_bit_cast(bf16x16, u);
}
__device__ __forceinline__ floatx8 bfwmma(bf16x16 a, bf16x16 b, floatx8 c) {
    return __builtin_amdgcn_wmma_f32_16x16x32_bf16(false, a, false, b,
                                                   (short)0, c, false, false);
}

// ---------------------------------------------------------------------------
// Prep: fp32 -> bf16 flat convert, and transposing convert for weights
// ---------------------------------------------------------------------------
__global__ void cvt_bf16(const float* __restrict__ in,
                         unsigned short* __restrict__ out, int n) {
    int i = blockIdx.x * blockDim.x + threadIdx.x;
    if (i < n) out[i] = f2bf(in[i]);
}

// in [K][N] row-major -> out [N][K] bf16
__global__ void transpose_cvt(const float* __restrict__ in,
                              unsigned short* __restrict__ out, int K, int N) {
    int i = blockIdx.x * blockDim.x + threadIdx.x;
    if (i < K * N) {
        int n = i % N, k = i / N;
        out[n * K + k] = f2bf(in[i]);
    }
}

// ---------------------------------------------------------------------------
// GEMM core: wave computes a 16x64 tile; A [M][K] bf16, Bt [N][K] bf16.
// Ping-pong double buffering (unroll-by-2, no register copies): while the
// WMMAs for chunk k run out of one buffer set, loads for k+32 fill the other.
// ---------------------------------------------------------------------------
__global__ __launch_bounds__(256) void gemm_qkv(const unsigned short* __restrict__ Abf,
                                                const unsigned short* __restrict__ Bt,
                                                const float* __restrict__ bias,
                                                unsigned short* __restrict__ out) {
    const int lane = threadIdx.x, wave = threadIdx.y;
    const int m0 = blockIdx.y * 16;
    const int n0 = (blockIdx.x * 8 + wave) * 64;
    const int kh = lane >> 4, l16 = lane & 15;
    const unsigned short* ap = Abf + (m0 + l16) * DMODEL + kh * 8;

    auto loadA = [&](int k0) { return frag(ld8(ap + k0), ld8(ap + k0 + 16)); };
    auto loadB = [&](int k0, int j) {
        const unsigned short* bp = Bt + (n0 + j * 16 + l16) * DMODEL + k0 + kh * 8;
        return frag(ld8(bp), ld8(bp + 16));
    };

    floatx8 acc[4] = {};
    bf16x16 aA, aB, bA[4], bB[4];

    // compute chunk held in (a_u,b_u); load chunk kload into (a_l,b_l)
    auto step = [&](bf16x16& a_u, bf16x16 (&b_u)[4],
                    bf16x16& a_l, bf16x16 (&b_l)[4], int kload) {
        a_l = loadA(kload);
#pragma unroll
        for (int j = 0; j < 4; ++j) b_l[j] = loadB(kload, j);
        __builtin_prefetch(ap + kload + 256, 0, 1);
#pragma unroll
        for (int j = 0; j < 4; ++j) acc[j] = bfwmma(a_u, b_u[j], acc[j]);
    };

    aA = loadA(0);
#pragma unroll
    for (int j = 0; j < 4; ++j) bA[j] = loadB(0, j);

    for (int k0 = 0; k0 < DMODEL; k0 += 64) {
        const int k1 = k0 + 32;                                    // < DMODEL
        const int k2 = (k0 + 64 < DMODEL) ? k0 + 64 : DMODEL - 32; // clamp tail
        step(aA, bA, aB, bB, k1);
        step(aB, bB, aA, bA, k2);
    }

#pragma unroll
    for (int j = 0; j < 4; ++j) {
        const int col = n0 + j * 16 + l16;
        const float bv = bias[col];
#pragma unroll
        for (int r = 0; r < 8; ++r)
            out[(m0 + r + 8 * kh) * N3 + col] = f2bf(acc[j][r] + bv);
    }
}

__global__ __launch_bounds__(256) void gemm_out(const unsigned short* __restrict__ Abf,
                                                const unsigned short* __restrict__ Bt,
                                                const float* __restrict__ bias,
                                                float* __restrict__ out) {
    const int lane = threadIdx.x, wave = threadIdx.y;
    const int m0 = blockIdx.y * 16;
    const int n0 = (blockIdx.x * 8 + wave) * 64;
    const int kh = lane >> 4, l16 = lane & 15;
    const unsigned short* ap = Abf + (m0 + l16) * DMODEL + kh * 8;

    auto loadA = [&](int k0) { return frag(ld8(ap + k0), ld8(ap + k0 + 16)); };
    auto loadB = [&](int k0, int j) {
        const unsigned short* bp = Bt + (n0 + j * 16 + l16) * DMODEL + k0 + kh * 8;
        return frag(ld8(bp), ld8(bp + 16));
    };

    floatx8 acc[4] = {};
    bf16x16 aA, aB, bA[4], bB[4];

    auto step = [&](bf16x16& a_u, bf16x16 (&b_u)[4],
                    bf16x16& a_l, bf16x16 (&b_l)[4], int kload) {
        a_l = loadA(kload);
#pragma unroll
        for (int j = 0; j < 4; ++j) b_l[j] = loadB(kload, j);
        __builtin_prefetch(ap + kload + 256, 0, 1);
#pragma unroll
        for (int j = 0; j < 4; ++j) acc[j] = bfwmma(a_u, b_u[j], acc[j]);
    };

    aA = loadA(0);
#pragma unroll
    for (int j = 0; j < 4; ++j) bA[j] = loadB(0, j);

    for (int k0 = 0; k0 < DMODEL; k0 += 64) {
        const int k1 = k0 + 32;
        const int k2 = (k0 + 64 < DMODEL) ? k0 + 64 : DMODEL - 32;
        step(aA, bA, aB, bB, k1);
        step(aB, bB, aA, bA, k2);
    }

#pragma unroll
    for (int j = 0; j < 4; ++j) {
        const int col = n0 + j * 16 + l16;
        const float bv = bias[col];
#pragma unroll
        for (int r = 0; r < 8; ++r)
            out[(m0 + r + 8 * kh) * DMODEL + col] = acc[j][r] + bv;
    }
}

// ---------------------------------------------------------------------------
// RoPE on q,k (1/8 scale folded into q); pack q,k as [b,h][s][d] and V
// TRANSPOSED as [b,h][d][s] so the P.V B-fragments are contiguous loads.
// ---------------------------------------------------------------------------
__global__ void rotary_pack(const unsigned short* __restrict__ qkv,
                            unsigned short* __restrict__ qp,
                            unsigned short* __restrict__ kp,
                            unsigned short* __restrict__ vt) {
    const int idx = blockIdx.x * blockDim.x + threadIdx.x; // b,h,s,d packed
    const int d = idx & 63;
    const int s = (idx >> 6) & (SEQ - 1);
    const int h = (idx >> 17) & (NHEAD - 1);
    const int b = idx >> 21;
    const int base = (s * BATCH + b) * N3 + h * (3 * HD);
    const int dp = d ^ 32;                                  // rotary partner

    const float invf  = __powf(10000.0f, -(float)(d & 31) * (1.0f / 32.0f));
    const float theta = (float)s * invf;
    float sn, cs;
    __sincosf(theta, &sn, &cs);

    const float q  = bf2f(qkv[base + d]);
    const float qP = bf2f(qkv[base + dp]);
    const float k  = bf2f(qkv[base + HD + d]);
    const float kP = bf2f(qkv[base + HD + dp]);
    const float rq = (d < 32) ? -qP : qP;
    const float rk = (d < 32) ? -kP : kP;

    const int bh = b * NHEAD + h;
    const int o = (bh * SEQ + s) * HD + d;
    qp[o] = f2bf((q * cs + rq * sn) * 0.125f);   // hd^-0.5 = 0.125
    kp[o] = f2bf(k * cs + rk * sn);
    vt[(bh * HD + d) * SEQ + s] = qkv[base + 2 * HD + d];
}

// ---------------------------------------------------------------------------
// Flash attention: one wave = one 16-row q tile. K-chunks of 32, ping-pong
// double-buffered K fragments (unroll-by-2, no copies); V loads hoisted so
// they overlap the score WMMAs + softmax.
// ---------------------------------------------------------------------------
__global__ __launch_bounds__(256) void flash_attn(const unsigned short* __restrict__ qp,
                                                  const unsigned short* __restrict__ kp,
                                                  const unsigned short* __restrict__ vt,
                                                  unsigned short* __restrict__ attn_out) {
    __shared__ unsigned short p_lds[8][16][32];   // per-wave P tile (bf16)

    const int lane = threadIdx.x;
    const int wave = threadIdx.y;
    const int tile = blockIdx.x * 8 + wave;       // 0..4095
    const int qt = tile & 127;                    // q tile within head
    const int bh = tile >> 7;                     // 0..31  (b*16 + h)
    const int kh  = lane >> 4;
    const int l16 = lane & 15;

    const unsigned short* qbase = qp + (size_t)bh * SEQ * HD;
    const unsigned short* kbase = kp + (size_t)bh * SEQ * HD;
    const unsigned short* vbase = vt + (size_t)bh * HD * SEQ;

    auto loadK = [&](int c0, int nt, int kk) {
        const unsigned short* kr = kbase + (c0 + nt * 16 + l16) * HD + kh * 8 + kk * 32;
        return frag(ld8(kr), ld8(kr + 16));
    };
    auto loadV = [&](int c0, int nt) {
        const unsigned short* vr = vbase + (nt * 16 + l16) * SEQ + c0 + kh * 8;
        return frag(ld8(vr), ld8(vr + 16));
    };

    // q A-fragments: kk=0 -> d 0..31, kk=1 -> d 32..63 (loaded once)
    bf16x16 qa[2];
    {
        const unsigned short* qr = qbase + (qt * 16 + l16) * HD + kh * 8;
        qa[0] = frag(ld8(qr), ld8(qr + 16));
        qa[1] = frag(ld8(qr + 32), ld8(qr + 48));
    }

    float   m_run[8], l_run[8];
    floatx8 Ov[4] = {};
#pragma unroll
    for (int r = 0; r < 8; ++r) { m_run[r] = -1e30f; l_run[r] = 0.0f; }

    bf16x16 kfA[2][2], kfB[2][2];
#pragma unroll
    for (int nt = 0; nt < 2; ++nt)
#pragma unroll
        for (int kk = 0; kk < 2; ++kk) kfA[nt][kk] = loadK(0, nt, kk);

    // process chunk c0 using kf_u; load chunk c0+32 (clamped) into kf_l
    auto body = [&](bf16x16 (&kf_u)[2][2], bf16x16 (&kf_l)[2][2], int c0) {
        const int cn = (c0 + 32 < SEQ) ? (c0 + 32) : c0;
#pragma unroll
        for (int nt = 0; nt < 2; ++nt)
#pragma unroll
            for (int kk = 0; kk < 2; ++kk) kf_l[nt][kk] = loadK(cn, nt, kk);
        bf16x16 vf[4];
#pragma unroll
        for (int nt = 0; nt < 4; ++nt) vf[nt] = loadV(c0, nt);

        // scores: q (16x64) @ kT (64x32)
        floatx8 sc[2];
#pragma unroll
        for (int nt = 0; nt < 2; ++nt) {
            floatx8 s = {};
            s = bfwmma(qa[0], kf_u[nt][0], s);
            s = bfwmma(qa[1], kf_u[nt][1], s);
            sc[nt] = s;
        }
        // online softmax (row = (r, lane-half); reduce across 16 lanes)
        float corr[8];
#pragma unroll
        for (int r = 0; r < 8; ++r) {
            float mx = fmaxf(sc[0][r], sc[1][r]);
            mx = fmaxf(mx, __shfl_xor(mx, 1));
            mx = fmaxf(mx, __shfl_xor(mx, 2));
            mx = fmaxf(mx, __shfl_xor(mx, 4));
            mx = fmaxf(mx, __shfl_xor(mx, 8));
            const float mnew = fmaxf(m_run[r], mx);
            const float cf = __expf(m_run[r] - mnew);
            const float p0 = __expf(sc[0][r] - mnew);
            const float p1 = __expf(sc[1][r] - mnew);
            float rs = p0 + p1;
            rs += __shfl_xor(rs, 1);
            rs += __shfl_xor(rs, 2);
            rs += __shfl_xor(rs, 4);
            rs += __shfl_xor(rs, 8);
            l_run[r] = l_run[r] * cf + rs;
            m_run[r] = mnew;
            corr[r]  = cf;
            sc[0][r] = p0;
            sc[1][r] = p1;
        }
#pragma unroll
        for (int nt = 0; nt < 4; ++nt)
#pragma unroll
            for (int r = 0; r < 8; ++r)
                Ov[nt][r] = Ov[nt][r] * corr[r];

        // relayout P (C-layout -> A-fragment) through LDS
#pragma unroll
        for (int nt = 0; nt < 2; ++nt)
#pragma unroll
            for (int r = 0; r < 8; ++r)
                p_lds[wave][r + 8 * kh][nt * 16 + l16] = f2bf(sc[nt][r]);
        // same-wave DS ops are in-order; compiler inserts s_wait_dscnt before use
        bf16x16 pa = frag(*(const ushort8*)&p_lds[wave][l16][kh * 8],
                          *(const ushort8*)&p_lds[wave][l16][kh * 8 + 16]);

        // O += P (16x32) @ V (32x64)
#pragma unroll
        for (int nt = 0; nt < 4; ++nt)
            Ov[nt] = bfwmma(pa, vf[nt], Ov[nt]);
    };

    for (int c0 = 0; c0 < SEQ; c0 += 64) {
        body(kfA, kfB, c0);
        body(kfB, kfA, c0 + 32);
    }

    // finalize: O / l -> attn[s, b, h*64 + d] (bf16)
    const int b = bh >> 4, h = bh & 15;
#pragma unroll
    for (int r = 0; r < 8; ++r) {
        const int   srow = qt * 16 + r + 8 * kh;
        const float invl = 1.0f / l_run[r];
        const size_t ob  = ((size_t)srow * BATCH + b) * DMODEL + h * HD;
#pragma unroll
        for (int nt = 0; nt < 4; ++nt)
            attn_out[ob + nt * 16 + l16] = f2bf(Ov[nt][r] * invl);
    }
}

// ---------------------------------------------------------------------------
extern "C" void kernel_launch(void* const* d_in, const int* in_sizes, int n_in,
                              void* d_out, int out_size, void* d_ws, size_t ws_size,
                              hipStream_t stream) {
    const float* x     = (const float*)d_in[0];   // [2048, 2, 1024]
    const float* W_qkv = (const float*)d_in[1];   // [1024, 3072]
    const float* b_qkv = (const float*)d_in[2];   // [3072]
    const float* W_out = (const float*)d_in[3];   // [1024, 1024]
    const float* b_out = (const float*)d_in[4];   // [1024]
    float* out = (float*)d_out;                   // [2048, 2, 1024]

    // workspace (bf16, 64 MB total):
    //  x_bf 8MB (reused as attn later) | Wqkv^T 6MB | Wout^T 2MB | qkv 24MB
    //  | q 8MB | k 8MB | v^T 8MB
    char* ws = (char*)d_ws;
    unsigned short* x_bf    = (unsigned short*)ws;
    unsigned short* attn_bf = x_bf;                               // reuse after flash
    unsigned short* Wqkv_t  = (unsigned short*)(ws + 8388608);
    unsigned short* Wout_t  = (unsigned short*)(ws + 8388608 + 6291456);
    unsigned short* qkv_bf  = (unsigned short*)(ws + 8388608 + 6291456 + 2097152);
    unsigned short* qpack   = qkv_bf + (size_t)SB * N3;
    unsigned short* kpack   = qpack + (size_t)BATCH * NHEAD * SEQ * HD;
    unsigned short* vtrans  = kpack + (size_t)BATCH * NHEAD * SEQ * HD;

    dim3 blk(32, 8);
    const int nx = SB * DMODEL;                 // 4,194,304
    // prep: convert + transpose (bandwidth-bound, ~48MB)
    cvt_bf16<<<nx / 256, 256, 0, stream>>>(x, x_bf, nx);
    transpose_cvt<<<(DMODEL * N3) / 256, 256, 0, stream>>>(W_qkv, Wqkv_t, DMODEL, N3);
    transpose_cvt<<<(DMODEL * DMODEL) / 256, 256, 0, stream>>>(W_out, Wout_t, DMODEL, DMODEL);
    // 1) QKV projection: 16x64 tile per wave
    gemm_qkv<<<dim3(N3 / 64 / 8, SB / 16), blk, 0, stream>>>(x_bf, Wqkv_t, b_qkv, qkv_bf);
    // 2) rotary + repack (+ V transpose)
    rotary_pack<<<nx / 256, 256, 0, stream>>>(qkv_bf, qpack, kpack, vtrans);
    // 3) flash attention: 4096 q tiles, 8 waves per block
    flash_attn<<<dim3(512), blk, 0, stream>>>(qpack, kpack, vtrans, attn_bf);
    // 4) output projection
    gemm_out<<<dim3(DMODEL / 64 / 8, SB / 16), blk, 0, stream>>>(attn_bf, Wout_t, b_out, out);
}
